// Attention_3367254360539
// MI455X (gfx1250) — matrix operand order repaired and checked by reading, not compiled, not run on hardware
//
#include <hip/hip_runtime.h>
#include <hip/hip_bf16.h>

// ---------------------------------------------------------------------------
// GPT-2 attention block on MI455X (gfx1250, wave32) via bf16 WMMA + flash attn
// Shapes fixed by the harness: B=4, S=2048, D=1024, H=16, hd=64.
//
// Roofline: ~137 GFLOP total; flash attention keeps the 1 GB of S x S score
// traffic out of HBM (~250 MB moved => ~11 us at 23.3 TB/s), so the matmuls
// ride the v_wmma_f32_16x16x32_bf16 pipe (fp32 accumulate).  Softmax row-sums
// are computed with WMMA against a ones-matrix to keep the DS pipe
// (ds_bpermute from __shfl_xor) off the critical path.
// ---------------------------------------------------------------------------

typedef __attribute__((ext_vector_type(16))) __bf16 v16bf;
typedef __attribute__((ext_vector_type(8)))  float  v8f;

struct alignas(16) U4 { unsigned int x, y, z, w; };
union Frag16 { v16bf v; U4 q[2]; unsigned short u[16]; };

__device__ __forceinline__ v8f wmma_bf16(v16bf a, v16bf b, v8f c) {
  // D = A(16x32 bf16) * B(32x16 bf16) + C(16x16 f32)
  return __builtin_amdgcn_wmma_f32_16x16x32_bf16(
      /*neg_a=*/false, a, /*neg_b=*/false, b,
      /*c_mod=*/(short)0, c, /*reuse_a=*/false, /*reuse_b=*/false);
}

// Load 16 bf16 -> B-fragment lane slice (K chunks at 8*lh and 16+8*lh; p = base + 16*lh)
__device__ __forceinline__ v16bf ldb16(const __bf16* p) {
  Frag16 t;
  const __bf16* q = p - 8 * ((threadIdx.x >> 4) & 1);
  t.q[0] = *(const U4*)q;
  t.q[1] = *(const U4*)(q + 16);
  return t.v;
}

__device__ __forceinline__ unsigned short bfbits(float f) {
  union { __bf16 h; unsigned short u; } t;
  t.h = (__bf16)f;
  return t.u;
}

// Row-max across the 16-lane halves (matches C-frag N striping).
__device__ __forceinline__ float hmax16(float v) {
#pragma unroll
  for (int m = 1; m < 16; m <<= 1) v = fmaxf(v, __shfl_xor(v, m, 32));
  return v;
}

// ---------------------------------------------------------------------------
// Elementwise fp32 -> bf16 convert
// ---------------------------------------------------------------------------
__global__ void cvt_bf16(const float* __restrict__ in, __bf16* __restrict__ out, int n) {
  for (int i = blockIdx.x * blockDim.x + threadIdx.x; i < n; i += gridDim.x * blockDim.x)
    out[i] = (__bf16)in[i];
}

// Transpose [K,N] fp32 -> [N,K] bf16 (so GEMM B-fragments are contiguous loads)
__global__ void transpose_cvt(const float* __restrict__ in, __bf16* __restrict__ out,
                              int K, int N) {
  const int total = K * N;
  for (int i = blockIdx.x * blockDim.x + threadIdx.x; i < total; i += gridDim.x * blockDim.x) {
    const int nrow = i / K;
    const int k    = i - nrow * K;
    out[i] = (__bf16)in[(size_t)k * N + nrow];
  }
}

// ---------------------------------------------------------------------------
// WMMA GEMM: C[M,N] = A_bf16[M,K] @ Bt_bf16[N,K]^T + bias
// Block = 256 threads = 8 waves (2 along M x 4 along N); wave tile 32x64.
// MODE 0: fp32 output.  MODE 1: fused head-split QKV epilogue -> q/k bf16
// [B,H,S,64] (q pre-scaled by 1/sqrt(64), exact in bf16) and vT [B,H,64,S].
// ---------------------------------------------------------------------------
template <int MODE>
__global__ __launch_bounds__(256) void gemm_wmma(
    const __bf16* __restrict__ A, const __bf16* __restrict__ Bt,
    const float* __restrict__ bias, float* __restrict__ Cf,
    __bf16* __restrict__ oq, __bf16* __restrict__ ok, __bf16* __restrict__ ovT,
    int M, int N, int K) {
  const int lane = threadIdx.x & 31;
  const int wave = __builtin_amdgcn_readfirstlane(threadIdx.x >> 5);  // scalar
  const int m  = lane & 15;
  const int lh = lane >> 4;
  const int m0 = blockIdx.y * 64  + (wave & 1) * 32;
  const int n0 = blockIdx.x * 256 + (wave >> 1) * 64;

  v8f c[2][4] = {};

  // A-fragment lane base: row (m0+m), K chunks at 8*lh and 16+8*lh
  const __bf16* a0p = A + (size_t)(m0 + m) * K + 8 * lh;
  const __bf16* a1p = a0p + (size_t)16 * K;

  for (int k0 = 0; k0 < K; k0 += 32) {
    Frag16 a0, a1;
    a0.q[0] = *(const U4*)(a0p + k0);
    a0.q[1] = *(const U4*)(a0p + k0 + 16);
    a1.q[0] = *(const U4*)(a1p + k0);
    a1.q[1] = *(const U4*)(a1p + k0 + 16);
    if (k0 + 32 < K) {
      __builtin_prefetch(a0p + k0 + 32, 0, 0);   // global_prefetch_b8
      __builtin_prefetch(a1p + k0 + 32, 0, 0);
    }
    v16bf bfr[4];
#pragma unroll
    for (int f = 0; f < 4; ++f)
      bfr[f] = ldb16(Bt + (size_t)(n0 + f * 16 + m) * K + k0 + 16 * lh);
#pragma unroll
    for (int f = 0; f < 4; ++f) {
      c[0][f] = wmma_bf16(a0.v, bfr[f], c[0][f]);
      c[1][f] = wmma_bf16(a1.v, bfr[f], c[1][f]);
    }
  }

#pragma unroll
  for (int i = 0; i < 2; ++i) {
#pragma unroll
    for (int f = 0; f < 4; ++f) {
      const int col = n0 + f * 16 + m;
      const float bsv = bias[col];
#pragma unroll
      for (int r = 0; r < 8; ++r) {
        const int row = m0 + i * 16 + lh * 8 + r;   // C-frag: M = r + 8*lh
        float v = c[i][f][r] + bsv;
        if (MODE == 0) {
          Cf[(size_t)row * N + col] = v;
        } else {
          const int which = col >> 10;          // 0=q, 1=k, 2=v  (N = 3*1024)
          const int hh    = (col >> 6) & 15;    // head
          const int d     = col & 63;           // dim in head
          const int bi    = row >> 11;          // batch (S = 2048)
          const int s     = row & 2047;
          const size_t bh = (size_t)(bi * 16 + hh);
          if (which == 0) {
            oq[(bh * 2048 + s) * 64 + d] = (__bf16)(v * 0.125f);  // 1/sqrt(hd)
          } else if (which == 1) {
            ok[(bh * 2048 + s) * 64 + d] = (__bf16)v;
          } else {
            ovT[(bh * 64 + d) * 2048 + s] = (__bf16)v;
          }
        }
      }
    }
  }
}

// ---------------------------------------------------------------------------
// Flash attention: block = 8 waves, each wave owns 16 query rows.
// Per 64-key block: 8 WMMA (scores) + online softmax + 2 WMMA (row-sum via
// ones-matrix) + 8 WMMA (P@V).  P is re-shaped C-frag -> A-frag through a
// padded per-wave LDS buffer (all b128 accesses 16B-aligned).
// ---------------------------------------------------------------------------
__global__ __launch_bounds__(256) void flash_attn(
    const __bf16* __restrict__ Q, const __bf16* __restrict__ Kk,
    const __bf16* __restrict__ Vt, __bf16* __restrict__ O) {
  __shared__ __align__(16) unsigned short pl[8][16 * 72];  // 16 rows x 72 bf16/wave

  const int lane = threadIdx.x & 31;
  const int wave = __builtin_amdgcn_readfirstlane(threadIdx.x >> 5);  // scalar:
  const int m  = lane & 15;      // -> q0, loop bound, LDS base become uniform,
  const int lh = lane >> 4;      //    killing per-lane EXEC-mask loop overhead
  const int h  = blockIdx.y;
  const int b  = blockIdx.z;
  const int q0 = blockIdx.x * 128 + wave * 16;

  const size_t bh = (size_t)(b * 16 + h);
  const __bf16* qh = Q  + bh * 2048 * 64;
  const __bf16* kh = Kk + bh * 2048 * 64;
  const __bf16* vh = Vt + bh * 64 * 2048;

  // Q A-fragments (dims 0..31 and 32..63), loaded once; Q is pre-scaled.
  Frag16 aq0, aq1;
  {
    const __bf16* qp = qh + (size_t)(q0 + m) * 64 + 8 * lh;
    aq0.q[0] = *(const U4*)qp;        aq0.q[1] = *(const U4*)(qp + 16);
    aq1.q[0] = *(const U4*)(qp + 32); aq1.q[1] = *(const U4*)(qp + 48);
  }

  // Ones B-fragment for WMMA row-sum of P.
  Frag16 ones;
#pragma unroll
  for (int i = 0; i < 16; ++i) ones.u[i] = 0x3F80;   // bf16 1.0

  v8f o[4] = {};
  v8f ls   = {};                 // softmax denominator as a C-fragment
  float mi[8];
#pragma unroll
  for (int r = 0; r < 8; ++r) mi[r] = -1e30f;

  unsigned short* pw = &pl[wave][0];

  // V row bases are kb-invariant: dim rows f*16+m of vT (stride S).
  const __bf16* vrow[4];
#pragma unroll
  for (int f = 0; f < 4; ++f) vrow[f] = vh + (size_t)(f * 16 + m) * 2048;

  for (int kb0 = 0; kb0 <= q0 + 15; kb0 += 64) {
    // ---- scores: S = Q @ K^T for keys [kb0, kb0+64) (already scaled) ----
    v8f s[4] = {};
#pragma unroll
    for (int j = 0; j < 4; ++j) {
      int key = kb0 + j * 16 + m;
      if (key > 2047) key = 2047;                 // clamp; masked anyway
      const __bf16* kp = kh + (size_t)key * 64 + 16 * lh;
      s[j] = wmma_bf16(aq0.v, ldb16(kp),      s[j]);
      s[j] = wmma_bf16(aq1.v, ldb16(kp + 32), s[j]);
    }

    // ---- causal mask (-10000) + online softmax (row-max via shuffles) ----
    const bool full = (kb0 + 63 <= q0);           // wave-uniform: no masking
#pragma unroll
    for (int r = 0; r < 8; ++r) {
      const int qi = q0 + lh * 8 + r;
      float v0 = s[0][r], v1 = s[1][r], v2 = s[2][r], v3 = s[3][r];
      if (!full) {
        if (kb0 + m      > qi) v0 = -10000.0f;
        if (kb0 + 16 + m > qi) v1 = -10000.0f;
        if (kb0 + 32 + m > qi) v2 = -10000.0f;
        if (kb0 + 48 + m > qi) v3 = -10000.0f;
      }
      const float bm = hmax16(fmaxf(fmaxf(v0, v1), fmaxf(v2, v3)));
      const float mn = fmaxf(mi[r], bm);
      const float al = __expf(mi[r] - mn);
      mi[r] = mn;
      const float p0 = __expf(v0 - mn);
      const float p1 = __expf(v1 - mn);
      const float p2 = __expf(v2 - mn);
      const float p3 = __expf(v3 - mn);
      o[0][r] *= al; o[1][r] *= al; o[2][r] *= al; o[3][r] *= al; ls[r] *= al;
      const int rr = (lh * 8 + r) * 72 + m;       // 144B rows (16B-aligned)
      pw[rr]      = bfbits(p0);
      pw[rr + 16] = bfbits(p1);
      pw[rr + 32] = bfbits(p2);
      pw[rr + 48] = bfbits(p3);
    }

    // ---- re-shape P (C-frag) -> two A-frags via LDS ----
    Frag16 ap0, ap1;
    {
      const unsigned short* pp = pw + m * 72 + 8 * lh;
      ap0.q[0] = *(const U4*)pp;        ap0.q[1] = *(const U4*)(pp + 16);
      ap1.q[0] = *(const U4*)(pp + 32); ap1.q[1] = *(const U4*)(pp + 48);
    }

    // ---- denominator: ls += P @ ones  (every column = row-sum of P) ----
    ls = wmma_bf16(ap0.v, ones.v, ls);
    ls = wmma_bf16(ap1.v, ones.v, ls);

    // ---- O += P @ V: batch 4 B-frag loads per key-half, then 4 WMMAs ----
    int kk0 = kb0 + 16 * lh;      if (kk0 > 2032) kk0 = 2032;
    int kk1 = kb0 + 32 + 16 * lh; if (kk1 > 2032) kk1 = 2032;
    {
      v16bf bv[4];
#pragma unroll
      for (int f = 0; f < 4; ++f) bv[f] = ldb16(vrow[f] + kk0);
#pragma unroll
      for (int f = 0; f < 4; ++f) o[f] = wmma_bf16(ap0.v, bv[f], o[f]);
#pragma unroll
      for (int f = 0; f < 4; ++f) bv[f] = ldb16(vrow[f] + kk1);
#pragma unroll
      for (int f = 0; f < 4; ++f) o[f] = wmma_bf16(ap1.v, bv[f], o[f]);
    }
  }

  // ---- normalize and write merged-head bf16 output [B,S,D] ----
#pragma unroll
  for (int r = 0; r < 8; ++r) {
    const float inv = 1.0f / ls[r];
    const size_t row = (size_t)b * 2048 + q0 + lh * 8 + r;
    __bf16* op = O + row * 1024 + h * 64 + m;
    op[0]  = (__bf16)(o[0][r] * inv);
    op[16] = (__bf16)(o[1][r] * inv);
    op[32] = (__bf16)(o[2][r] * inv);
    op[48] = (__bf16)(o[3][r] * inv);
  }
}

// ---------------------------------------------------------------------------
// Host-side orchestration
// ---------------------------------------------------------------------------
extern "C" void kernel_launch(void* const* d_in, const int* in_sizes, int n_in,
                              void* d_out, int out_size, void* d_ws, size_t ws_size,
                              hipStream_t stream) {
  (void)in_sizes; (void)n_in; (void)out_size; (void)ws_size;
  const float* x      = (const float*)d_in[0];   // [4,2048,1024]
  const float* attn_w = (const float*)d_in[1];   // [1024,3072]
  const float* attn_b = (const float*)d_in[2];   // [3072]
  const float* proj_w = (const float*)d_in[3];   // [1024,1024]
  const float* proj_b = (const float*)d_in[4];   // [1024]
  float* out = (float*)d_out;                    // [4,2048,1024] fp32

  char* ws = (char*)d_ws;                        // ~92 MB total, 256B-aligned slots
  __bf16* x_bf   = (__bf16*)(ws);                //  8,388,608 el -> 16,777,216 B
  __bf16* wqkvT  = (__bf16*)(ws + 16777216);     //  [3072,1024] ->  6,291,456 B
  __bf16* wprojT = (__bf16*)(ws + 23068672);     //  [1024,1024] ->  2,097,152 B
  __bf16* qb     = (__bf16*)(ws + 25165824);     //  [4,16,2048,64] -> 16,777,216 B
  __bf16* kb     = (__bf16*)(ws + 41943040);     //  [4,16,2048,64] -> 16,777,216 B
  __bf16* vT     = (__bf16*)(ws + 58720256);     //  [4,16,64,2048] -> 16,777,216 B
  __bf16* abf    = (__bf16*)(ws + 75497472);     //  [4,2048,1024]  -> 16,777,216 B

  const int M = 4 * 2048;   // 8192 rows
  const int D = 1024;

  cvt_bf16<<<4096, 256, 0, stream>>>(x, x_bf, M * D);
  transpose_cvt<<<4096, 256, 0, stream>>>(attn_w, wqkvT, D, 3 * D);
  transpose_cvt<<<2048, 256, 0, stream>>>(proj_w, wprojT, D, D);

  // QKV projection with fused head-split (+ q-scale) epilogue
  dim3 g1(3 * D / 256, M / 64);                  // (12, 128)
  gemm_wmma<1><<<g1, 256, 0, stream>>>(x_bf, wqkvT, attn_b, nullptr,
                                       qb, kb, vT, M, 3 * D, D);

  // Flash attention
  dim3 g2(2048 / 128, 16, 4);                    // (16, 16, 4)
  flash_attn<<<g2, 256, 0, stream>>>(qb, kb, vT, abf);

  // Output projection -> fp32 result
  dim3 g3(D / 256, M / 64);                      // (4, 128)
  gemm_wmma<0><<<g3, 256, 0, stream>>>(abf, wprojT, proj_b, out,
                                       nullptr, nullptr, nullptr, M, D, D);
}
